// LSTMClassifier_57887569215605
// MI455X (gfx1250) — matrix operand order repaired and checked by reading, not compiled, not run on hardware
//
#include <hip/hip_runtime.h>
#include <hip/hip_bf16.h>

// ---------------------------------------------------------------------------
// LSTM classifier forward for MI455X (gfx1250, wave32, WMMA, TDM).
//   Phase 0: prep    — cvt embedding to f16; transpose+cvt W_ih/W_hh to f16
//                      [K,4H]; zero h ping-pong; lengths; barrier init.
//   Phase 1: inproj  — x_proj[t,b,g] = emb_f16[x[b,t]]·W_ih^T + b_ih + b_hh
//                      (v_wmma_f32_16x16x32_f16, all-f16 operand loads)
//   Phase 2: scan    — persistent 16-WG kernel; W_hh^T slab staged into LDS
//                      by the Tensor Data Mover (tensor_load_to_lds + pad);
//                      c slice in LDS; h ping-pong in global (L2);
//                      one device-wide barrier per step.
//   Phase 3: logits  — last_h · fc_W^T + fc_b
// Workspace: ~4.2 MB + emb_f16 (32 MB) + x_proj (512 MB)  => ~0.58 GB.
// ---------------------------------------------------------------------------

#define T_SEQ  2048
#define B_SZ   32
#define VOCAB  32000
#define D_E    512
#define D_H    512
#define G4     2048      // 4*D_H
#define NWG    16        // persistent workgroups in scan kernel
#define HCHUNK 32        // hidden units owned per workgroup (512/16)

typedef __attribute__((ext_vector_type(16))) _Float16 v16h;
typedef __attribute__((ext_vector_type(8)))  _Float16 v8h;
typedef __attribute__((ext_vector_type(8)))  float    v8f;
typedef __attribute__((ext_vector_type(4)))  unsigned u32x4;
typedef __attribute__((ext_vector_type(8)))  int      i32x8;
typedef __attribute__((ext_vector_type(4)))  int      i32x4;

union V16H { v16h v; v8h h[2]; };

// ---- WMMA fragment loaders (layouts per CDNA5 ISA 7.12.2) ------------------
// A (16x32 f16): lanes 0-15 -> M=lane, K = {0..7, 16..23}; lanes 16-31 -> K+8.
__device__ __forceinline__ v16h load_a_row(const _Float16* row, int k0, int lane) {
  const int hsel = (lane >> 4) & 1;
  const _Float16* p = row + k0 + hsel * 8;
  V16H u;
  u.h[0] = *(const v8h*)(p);        // K chunk base
  u.h[1] = *(const v8h*)(p + 16);   // K chunk +16
  return u.v;
}
__device__ __forceinline__ v16h load_a_f16(const _Float16* base, int ld,
                                           int m0, int k0, int lane) {
  return load_a_row(base + (size_t)(m0 + (lane & 15)) * ld, k0, lane);
}

// B (32x16 f16), source pre-transposed to [K, N] row-major: lanes 0-15 ->
// N=lane, K = {0..7, 16..23}; lanes 16-31 -> K+8 (strided gather).
__device__ __forceinline__ v16h load_b_f16(const _Float16* bt, int ld,
                                           int k0, int n0, int lane) {
  const _Float16* p = bt + (size_t)(k0 + ((lane >> 4) & 1) * 8) * ld + n0 + (lane & 15);
  v16h b;
#pragma unroll
  for (int s = 0; s < 8; ++s) {
    b[s]     = p[(size_t)s * ld];
    b[8 + s] = p[(size_t)(16 + s) * ld];
  }
  return b;
}

__device__ __forceinline__ v8f wmma16(v16h a, v16h b, v8f c) {
  return __builtin_amdgcn_wmma_f32_16x16x32_f16(false, a, false, b,
                                                (short)0, c, false, false);
}

__device__ __forceinline__ float sigmoidf_(float v) {
  return 1.0f / (1.0f + __expf(-v));
}

// ---- Tensor Data Mover: 2D f16 tile -> LDS with row padding ----------------
// D# per CDNA5 ISA ch.8: group0 = {flags, lds_addr, global_addr, type=2},
// group1 = {mask/data_size/pad, tensor dims, tile dims, dim0 stride}.
__device__ __forceinline__ void tdm_load_2d_f16(
    unsigned lds_byte, unsigned long long gaddr,
    unsigned tensor_d0, unsigned tensor_d1,
    unsigned tile_d0, unsigned tile_d1,
    unsigned long long d0_stride,
    unsigned pad_interval, unsigned pad_amount, bool pad_en) {
  u32x4 g0;
  g0[0] = 1u;                                            // count=1, user mode
  g0[1] = lds_byte;                                      // lds_addr
  g0[2] = (unsigned)(gaddr & 0xffffffffull);             // global_addr lo
  g0[3] = (unsigned)((gaddr >> 32) & 0x1ffffffull)       // global_addr hi
        | (2u << 30);                                    // type = 2 ("image")
  i32x8 g1;
  unsigned w0 = (1u << 16);                              // data_size = 2 bytes
  if (pad_en) w0 |= (1u << 20) | (pad_interval << 22) | (pad_amount << 25);
  g1[0] = (int)w0;
  g1[1] = (int)((tensor_d0 & 0xffffu) << 16);            // abar=0 | dim0 lo16
  g1[2] = (int)(((tensor_d0 >> 16) & 0xffffu) | ((tensor_d1 & 0xffffu) << 16));
  g1[3] = (int)(((tensor_d1 >> 16) & 0xffffu) | ((tile_d0 & 0xffffu) << 16));
  g1[4] = (int)(tile_d1 & 0xffffu);                      // tile_dim2 = 0 (2D)
  g1[5] = (int)(unsigned)(d0_stride & 0xffffffffull);
  g1[6] = (int)(unsigned)((d0_stride >> 32) & 0xffffull);// dim1_stride = 0
  g1[7] = 0;
  i32x4 z4 = {0, 0, 0, 0};
#if __has_builtin(__builtin_amdgcn_tensor_load_to_lds)
#if defined(__clang_major__) && (__clang_major__ >= 23)
  i32x8 z8 = {0, 0, 0, 0, 0, 0, 0, 0};
  __builtin_amdgcn_tensor_load_to_lds(g0, g1, z4, z4, z8, 0);
#else
  __builtin_amdgcn_tensor_load_to_lds(g0, g1, z4, z4, 0);
#endif
#else
  (void)g0; (void)g1; (void)z4;
#endif
}

__device__ __forceinline__ void wait_tensorcnt0() {
#if __has_builtin(__builtin_amdgcn_s_wait_tensorcnt)
  __builtin_amdgcn_s_wait_tensorcnt(0);
#else
  asm volatile("s_wait_tensorcnt 0x0" ::: "memory");
#endif
}

// ---------------------------------------------------------------------------
// Phase 0: embedding f32->f16, weight transpose+cvt, h zero, lengths, barrier.
// ---------------------------------------------------------------------------
__global__ void lstm_prep(const float* __restrict__ emb,
                          const float* __restrict__ W_ih,
                          const float* __restrict__ W_hh,
                          const int*   __restrict__ attn,
                          _Float16* __restrict__ embH,
                          _Float16* __restrict__ WihT,
                          _Float16* __restrict__ WhhT,
                          _Float16* __restrict__ hbuf,
                          int* __restrict__ lengths,
                          unsigned* __restrict__ bar) {
  const size_t gid  = (size_t)blockIdx.x * blockDim.x + threadIdx.x;
  const size_t nthr = (size_t)gridDim.x * blockDim.x;
  const size_t NW = (size_t)D_E * G4;
  for (size_t i = gid; i < NW; i += nthr) {
    const size_t e = i / G4, g = i % G4;
    WihT[e * G4 + g] = (_Float16)W_ih[g * D_E + e];
    WhhT[e * G4 + g] = (_Float16)W_hh[g * D_H + e];
  }
  const size_t NE = (size_t)VOCAB * D_E;
  for (size_t i = gid; i < NE; i += nthr)
    embH[i] = (_Float16)emb[i];
  for (size_t i = gid; i < (size_t)2 * B_SZ * D_H; i += nthr)
    hbuf[i] = (_Float16)0.0f;
  if (gid < B_SZ) {
    int s = 0;
    for (int j = 0; j < T_SEQ; ++j) s += attn[gid * T_SEQ + j];
    lengths[gid] = s - 1;
  }
  if (gid == 0) bar[0] = 0u;
}

// ---------------------------------------------------------------------------
// Phase 1: fused embedding gather + input projection GEMM (all-f16 operands).
//   grid = (T, 16); WG = 256 thr = 8 waves; wave -> one 16-col N tile,
//   both 16-row M tiles (batch 32); K = 512 in steps of 32.
// ---------------------------------------------------------------------------
__global__ __launch_bounds__(256) void lstm_inproj_wmma(
    const long long* __restrict__ x, const _Float16* __restrict__ embH,
    const _Float16* __restrict__ WihT, const float* __restrict__ b_ih,
    const float* __restrict__ b_hh, float* __restrict__ xproj) {
  const int t    = blockIdx.x;
  const int wave = threadIdx.x >> 5;
  const int lane = threadIdx.x & 31;
  const int n0   = blockIdx.y * 128 + wave * 16;

  const int row = lane & 15;
  const long long id0 = x[(size_t)row        * T_SEQ + t];
  const long long id1 = x[(size_t)(row + 16) * T_SEQ + t];
  const _Float16* e0 = embH + (size_t)id0 * D_E;
  const _Float16* e1 = embH + (size_t)id1 * D_E;

  v8f acc0 = {};
  v8f acc1 = {};
#pragma unroll 4
  for (int k0 = 0; k0 < D_E; k0 += 32) {
    if (k0 + 32 < D_E)  // global_prefetch_b8 on next B slab
      __builtin_prefetch(WihT + (size_t)(k0 + 32 + lane) * G4 + n0, 0, 1);
    const v16h bf = load_b_f16(WihT, G4, k0, n0, lane);
    const v16h a0 = load_a_row(e0, k0, lane);
    const v16h a1 = load_a_row(e1, k0, lane);
    acc0 = wmma16(a0, bf, acc0);
    acc1 = wmma16(a1, bf, acc1);
  }

  const int   n    = n0 + (lane & 15);
  const float bias = b_ih[n] + b_hh[n];   // fold both biases here
  float* out = xproj + (size_t)t * B_SZ * G4;
  const int mb = (lane >> 4) ? 8 : 0;
#pragma unroll
  for (int v = 0; v < 8; ++v) {
    out[(size_t)(mb + v)      * G4 + n] = acc0[v] + bias;
    out[(size_t)(16 + mb + v) * G4 + n] = acc1[v] + bias;
  }
}

// ---------------------------------------------------------------------------
// Phase 2: persistent sequential scan. 16 WGs, each owns 32 hidden units
// (gate cols {i,f,g,o} x 32 = 128 local cols). W_hh^T slab (512x128 f16 =
// 128 KB) is DMA'd into LDS by the TDM with LDS row padding (64 B data +
// 192 B pad = 256 B rows); c slice lives in LDS; h ping-pongs in global.
// ---------------------------------------------------------------------------
__global__ __launch_bounds__(256) void lstm_scan_wmma(
    const float* __restrict__ xproj, const _Float16* __restrict__ WhhT,
    const int* __restrict__ lengths, _Float16* __restrict__ hbuf,
    float* __restrict__ last_h, unsigned* __restrict__ bar) {
  extern __shared__ __align__(16) char smem[];
  _Float16* whh  = (_Float16*)smem;                                  // 128 KB
  float*    gbuf = (float*)(smem + 512 * 128 * 2);                   // 16 KB
  float*    cbuf = (float*)(smem + 512 * 128 * 2 + 32 * 128 * 4);    // 4 KB

  const int wg   = blockIdx.x;
  const int tid  = threadIdx.x;
  const int wave = tid >> 5;
  const int lane = tid & 31;

#if __has_builtin(__builtin_amdgcn_tensor_load_to_lds)
  // TDM: one 2D tile per gate block. Tile = 512 rows x 32 f16 cols out of
  // WhhT[512][2048]; each 64-B row lands at LDS stride 256 B (pad 48 DWORDs
  // after every 16 DWORDs), interleaving the 4 blocks into whh[k][blk*32+u].
  if (wave == 0) {
#pragma unroll
    for (int blk = 0; blk < 4; ++blk) {
      const unsigned long long gaddr =
          (unsigned long long)(uintptr_t)(WhhT + blk * D_H + wg * HCHUNK);
      const unsigned ldsb = (unsigned)(uintptr_t)whh + blk * 64u;
      tdm_load_2d_f16(ldsb, gaddr,
                      /*tensor_d0=*/G4, /*tensor_d1=*/D_H,
                      /*tile_d0=*/HCHUNK, /*tile_d1=*/D_H,
                      /*d0_stride=*/G4,
                      /*pad_interval(16 dw)=*/3u, /*pad_amount(48 dw)=*/47u,
                      /*pad_en=*/true);
    }
    wait_tensorcnt0();
  }
#else
  for (int i = tid; i < 512 * 128; i += 256) {
    const int k = i >> 7, c = i & 127;
    whh[i] = WhhT[(size_t)k * G4 + (c >> 5) * D_H + wg * HCHUNK + (c & 31)];
  }
#endif
  for (int i = tid; i < B_SZ * HCHUNK; i += 256) cbuf[i] = 0.0f;
  __syncthreads();

  const int nloc = wave * 16;                 // local col tile for this wave
  const int lc   = nloc + (lane & 15);        // local col 0..127
  const int gcol = (lc >> 5) * D_H + wg * HCHUNK + (lc & 31);
  const int mb   = (lane >> 4) ? 8 : 0;

  // Each thread owns 4 fixed (m,u) cells; hoist their lengths out of the loop.
  int len_[4];
#pragma unroll
  for (int r = 0; r < 4; ++r) len_[r] = lengths[(tid + 256 * r) >> 5];

  for (int t = 0; t < T_SEQ; ++t) {
    const _Float16* hin  = hbuf + (size_t)(t & 1)       * B_SZ * D_H;
    _Float16*       hout = hbuf + (size_t)((t + 1) & 1) * B_SZ * D_H;
    const float*    xp   = xproj + (size_t)t * B_SZ * G4;

    // C init = precomputed input projection (+ both biases already folded)
    v8f acc0, acc1;
#pragma unroll
    for (int v = 0; v < 8; ++v) {
      acc0[v] = xp[(size_t)(mb + v)      * G4 + gcol];
      acc1[v] = xp[(size_t)(16 + mb + v) * G4 + gcol];
    }
    // gates += h(t-1) x W_hh^T  (A from global L2, B from LDS)
#pragma unroll 4
    for (int k0 = 0; k0 < D_H; k0 += 32) {
      const v16h bf = load_b_f16(whh, 128, k0, nloc, lane);
      const v16h a0 = load_a_f16(hin, D_H, 0,  k0, lane);
      const v16h a1 = load_a_f16(hin, D_H, 16, k0, lane);
      acc0 = wmma16(a0, bf, acc0);
      acc1 = wmma16(a1, bf, acc1);
    }
    // Publish pre-activations so each thread can pick up its (i,f,g,o).
#pragma unroll
    for (int v = 0; v < 8; ++v) {
      gbuf[(mb + v)      * 128 + lc] = acc0[v];
      gbuf[(16 + mb + v) * 128 + lc] = acc1[v];
    }
    __syncthreads();

    // Elementwise LSTM cell for the 32x32 (batch x unit) slice.
#pragma unroll
    for (int r = 0; r < 4; ++r) {
      const int i = tid + 256 * r;
      const int m = i >> 5, u = i & 31;
      const float gi = gbuf[m * 128 + u];
      const float gf = gbuf[m * 128 + 32 + u];
      const float gg = gbuf[m * 128 + 64 + u];
      const float go = gbuf[m * 128 + 96 + u];
      float c = cbuf[i];
      c = sigmoidf_(gf) * c + sigmoidf_(gi) * tanhf(gg);
      const float h = sigmoidf_(go) * tanhf(c);
      cbuf[i] = c;
      hout[(size_t)m * D_H + wg * HCHUNK + u] = (_Float16)h;
      if (t == len_[r])
        last_h[(size_t)m * D_H + wg * HCHUNK + u] = h;
    }

    // Device-wide barrier: monotonic arrival counter, target = NWG*(t+1).
    __threadfence();
    __syncthreads();
    if (tid == 0) {
      atomicAdd(bar, 1u);
      const unsigned target = (unsigned)NWG * (unsigned)(t + 1);
      while (atomicAdd(bar, 0u) < target) __builtin_amdgcn_s_sleep(2);
    }
    __syncthreads();
  }
}

// ---------------------------------------------------------------------------
// Phase 3: logits = last_h . fc_W^T + fc_b   (32x4 outputs, one thread each)
// ---------------------------------------------------------------------------
__global__ __launch_bounds__(128) void lstm_logits(
    const float* __restrict__ last_h, const float* __restrict__ fc_W,
    const float* __restrict__ fc_b, float* __restrict__ out) {
  const int i = threadIdx.x;        // 0..127
  if (i >= B_SZ * 4) return;
  const int m = i >> 2, c = i & 3;
  float s = fc_b[c];
  for (int j = 0; j < D_H; ++j)
    s += last_h[(size_t)m * D_H + j] * fc_W[c * D_H + j];
  out[m * 4 + c] = s;
}

// ---------------------------------------------------------------------------
// Workspace layout (bytes)
// ---------------------------------------------------------------------------
#define OFF_BAR  0u
#define OFF_LEN  256u
#define OFF_H    4096u                                 // 2*32*512 f16 = 64 KB
#define OFF_LH   (OFF_H + 65536u)                      // 32*512 f32  = 64 KB
#define OFF_WIH  (OFF_LH + 65536u)                     // 512*2048 f16 = 2 MB
#define OFF_WHH  (OFF_WIH + 2097152u)                  // 512*2048 f16 = 2 MB
#define OFF_EMB  (OFF_WHH + 2097152u)                  // 32000*512 f16 = 32 MB
#define OFF_XP   (OFF_EMB + 32768000u)                 // T*B*4H f32 = 512 MB

extern "C" void kernel_launch(void* const* d_in, const int* in_sizes, int n_in,
                              void* d_out, int out_size, void* d_ws, size_t ws_size,
                              hipStream_t stream) {
  const long long* x    = (const long long*)d_in[0];   // int64 [B,T]
  const int*       attn = (const int*)d_in[1];         // int32 [B,T]
  const float*     emb  = (const float*)d_in[2];       // [VOCAB, E]
  const float*     W_ih = (const float*)d_in[3];       // [4H, E]
  const float*     W_hh = (const float*)d_in[4];       // [4H, H]
  const float*     b_ih = (const float*)d_in[5];       // [4H]
  const float*     b_hh = (const float*)d_in[6];       // [4H]
  const float*     fc_W = (const float*)d_in[7];       // [4, H]
  const float*     fc_b = (const float*)d_in[8];       // [4]
  (void)in_sizes; (void)n_in; (void)out_size; (void)ws_size;

  char* ws = (char*)d_ws;
  unsigned* bar    = (unsigned*)(ws + OFF_BAR);
  int*      length = (int*)(ws + OFF_LEN);
  _Float16* hbuf   = (_Float16*)(ws + OFF_H);
  float*    last_h = (float*)(ws + OFF_LH);
  _Float16* WihT   = (_Float16*)(ws + OFF_WIH);
  _Float16* WhhT   = (_Float16*)(ws + OFF_WHH);
  _Float16* embH   = (_Float16*)(ws + OFF_EMB);
  float*    xproj  = (float*)(ws + OFF_XP);
  float*    out    = (float*)d_out;

  hipLaunchKernelGGL(lstm_prep, dim3(1024), dim3(256), 0, stream,
                     emb, W_ih, W_hh, attn, embH, WihT, WhhT, hbuf, length, bar);

  hipLaunchKernelGGL(lstm_inproj_wmma, dim3(T_SEQ, 16), dim3(256), 0, stream,
                     x, embH, WihT, b_ih, b_hh, xproj);

  const size_t smem = 512 * 128 * 2 + 32 * 128 * 4 + 32 * 32 * 4;  // 148 KB
  hipLaunchKernelGGL(lstm_scan_wmma, dim3(NWG), dim3(256), smem, stream,
                     xproj, WhhT, length, hbuf, last_h, bar);

  hipLaunchKernelGGL(lstm_logits, dim3(1), dim3(128), 0, stream,
                     last_h, fc_W, fc_b, out);
}
